// CVQVAE_51668456571490
// MI455X (gfx1250) — matrix operand order repaired and checked by reading, not compiled, not run on hardware
//
#include <hip/hip_runtime.h>

// CDNA5 / gfx1250 vector types for WMMA
typedef __attribute__((ext_vector_type(16))) _Float16 v16h;
typedef __attribute__((ext_vector_type(8)))  float    v8f;

#define SELU_SCALE 1.0507009873554805f
#define SELU_ALPHA 1.6732632423543772f

__device__ __forceinline__ float selu_f(float x) {
    return SELU_SCALE * (x > 0.f ? x : SELU_ALPHA * (__expf(x) - 1.f));
}

// ---------------------------------------------------------------------------
// Encoder conv: VALID 3x3, stride 1. Optional SELU, optional NHWC output
// (used by conv3 so the VQ kernel sees contiguous 16-float latent rows).
// ---------------------------------------------------------------------------
template <int CI, int CO, bool ACT, bool NHWC_OUT>
__global__ void enc_conv(const float* __restrict__ in, const float* __restrict__ w,
                         const float* __restrict__ bias, float* __restrict__ out,
                         int Bn, int Hi, int Wi) {
    const int Ho = Hi - 2, Wo = Wi - 2;
    const int total = Bn * CO * Ho * Wo;
    int t = blockIdx.x * blockDim.x + threadIdx.x;
    if (t >= total) return;
    int xq = t % Wo;
    int yq = (t / Wo) % Ho;
    int o  = (t / (Wo * Ho)) % CO;
    int b  = t / (Wo * Ho * CO);
    float acc = bias[o];
    const float* wp = w + o * CI * 9;
#pragma unroll
    for (int i = 0; i < CI; ++i) {
        const float* ip = in + ((size_t)(b * CI + i) * Hi + yq) * Wi + xq;
#pragma unroll
        for (int r = 0; r < 3; ++r)
#pragma unroll
            for (int s = 0; s < 3; ++s)
                acc = fmaf(ip[r * Wi + s], wp[i * 9 + r * 3 + s], acc);
    }
    if (ACT) acc = selu_f(acc);
    if (NHWC_OUT)
        out[((size_t)(b * Ho + yq) * Wo + xq) * CO + o] = acc;
    else
        out[((size_t)(b * CO + o) * Ho + yq) * Wo + xq] = acc;
}

// ---------------------------------------------------------------------------
// Transpose conv (stride 1, k=3, pad 2 full conv):
//   y[b][o][p][q] = db[o] + sum_{i,u,v} x[b][i][p-u][q-v] * w[i][o][u][v]
// w shape (CI, CO, 3, 3)
// ---------------------------------------------------------------------------
template <int CI, int CO, bool ACT>
__global__ void dec_conv(const float* __restrict__ in, const float* __restrict__ w,
                         const float* __restrict__ bias, float* __restrict__ out,
                         int Bn, int Hi, int Wi) {
    const int Ho = Hi + 2, Wo = Wi + 2;
    const int total = Bn * CO * Ho * Wo;
    int t = blockIdx.x * blockDim.x + threadIdx.x;
    if (t >= total) return;
    int q = t % Wo;
    int p = (t / Wo) % Ho;
    int o = (t / (Wo * Ho)) % CO;
    int b = t / (Wo * Ho * CO);
    float acc = bias[o];
#pragma unroll
    for (int i = 0; i < CI; ++i) {
        const float* ip = in + (size_t)(b * CI + i) * Hi * Wi;
        const float* wp = w + (i * CO + o) * 9;
#pragma unroll
        for (int u = 0; u < 3; ++u) {
            int yy = p - u;
            if (yy < 0 || yy >= Hi) continue;
#pragma unroll
            for (int v = 0; v < 3; ++v) {
                int xx = q - v;
                if (xx < 0 || xx >= Wi) continue;
                acc = fmaf(ip[(size_t)yy * Wi + xx], wp[u * 3 + v], acc);
            }
        }
    }
    if (ACT) acc = selu_f(acc);
    out[t] = acc;
}

// ---------------------------------------------------------------------------
// Codebook prep: per-row squared norm (f32, exact) + hi/lo f16 decomposition
// for the Markidis-style split GEMM.
// ---------------------------------------------------------------------------
__global__ void cb_prep(const float* __restrict__ cb, float* __restrict__ cbn,
                        _Float16* __restrict__ cbh, _Float16* __restrict__ cbl, int Kn) {
    int j = blockIdx.x * blockDim.x + threadIdx.x;
    if (j >= Kn) return;
    const float* r = cb + (size_t)j * 16;
    float nrm = 0.f;
#pragma unroll
    for (int d = 0; d < 16; ++d) {
        float v = r[d];
        nrm = fmaf(v, v, nrm);
        _Float16 h = (_Float16)v;
        cbh[(size_t)j * 16 + d] = h;
        cbl[(size_t)j * 16 + d] = (_Float16)(v - (float)h);
    }
    cbn[j] = nrm;
}

// ---------------------------------------------------------------------------
// Async global -> LDS staging helpers (CDNA5 ASYNCcnt path; inline asm since
// the builtins are not confirmed on this toolchain).
// ---------------------------------------------------------------------------
__device__ __forceinline__ void async_g2lds_b128(unsigned lds_byte, unsigned long long gaddr) {
    asm volatile("global_load_async_to_lds_b128 %0, %1, off"
                 :: "v"(lds_byte), "v"(gaddr) : "memory");
}
__device__ __forceinline__ void async_g2lds_b32(unsigned lds_byte, unsigned long long gaddr) {
    asm volatile("global_load_async_to_lds_b32 %0, %1, off"
                 :: "v"(lds_byte), "v"(gaddr) : "memory");
}
__device__ __forceinline__ void wait_asynccnt0() {
    asm volatile("s_wait_asynccnt 0x0" ::: "memory");
}
__device__ __forceinline__ unsigned lds_off(const void* p) {
    return (unsigned)(uintptr_t)p;   // low 32 bits of flat addr = LDS offset
}

// ---------------------------------------------------------------------------
// VQ argmin. Block = 256 threads = 8 waves; each wave owns one 16-row tile
// of zf. The 8 waves share the codebook stream: 128-entry chunks of the hi/lo
// f16 codebook (+ norms) are double-buffered in LDS via async global->LDS
// copies (ASYNCcnt), cutting L2 traffic 8x and overlapping copy with compute.
//
// Per 16x16 score tile: dot = z.e via two v_wmma_f32_16x16x32_f16 (hi/lo
// split packed in K: WMMA1 A=[z_hi|z_lo] B=[cb_hi|cb_hi], WMMA2 A=[z_hi|0]
// B=[cb_lo|.]), score = ||e||^2 - 2*dot, branchless running argmin,
// xor-shuffle reduction per half-wave. All branches are wave-uniform so EXEC
// stays all-ones around the WMMAs.
// ---------------------------------------------------------------------------
#define VQ_CHUNK 128   // codebook entries staged per round

__global__ void __launch_bounds__(256)
vq_argmin(const float* __restrict__ zf, const _Float16* __restrict__ cbh,
          const _Float16* __restrict__ cbl, const float* __restrict__ cbn,
          int* __restrict__ idx_out, int ntiles, int Kn) {
    __shared__ __align__(16) _Float16 sh_h[2][VQ_CHUNK * 16];
    __shared__ __align__(16) _Float16 sh_l[2][VQ_CHUNK * 16];
    __shared__ __align__(16) float    sh_n[2][VQ_CHUNK];

    const int tid    = threadIdx.x;
    const int lane   = tid & 31;
    const int wave_g = (int)((blockIdx.x * blockDim.x + tid) >> 5);
    const int tile   = wave_g < ntiles ? wave_g : ntiles - 1;  // clamp, no early exit

    const int l    = lane & 15;
    const int half = lane >> 4;
    const int row0 = tile * 16;
    const int kb   = half * 8;            // this lane's K-base within D=16

    // Build A tiles once per wave: a1 = [z_hi | z_lo], a2 = [z_hi | 0]
    const float* zrow = zf + (size_t)(row0 + l) * 16;
    v16h a1, a2;
#pragma unroll
    for (int j = 0; j < 8; ++j) {
        float z = zrow[kb + j];
        _Float16 h = (_Float16)z;
        a1[j]     = h;
        a2[j]     = h;
        a1[j + 8] = (_Float16)(z - (float)h);
        a2[j + 8] = (_Float16)0.f;
    }

    float bestv[8];
    int   besti[8];
#pragma unroll
    for (int r = 0; r < 8; ++r) { bestv[r] = 3.4e38f; besti[r] = 0; }

    const int nchunk = Kn / VQ_CHUNK;     // 64

    // Stage chunk 0 into buffer 0 (one b128 per thread per array, b32 norms
    // from waves 0-3; all conditions wave-uniform).
    {
        unsigned long long gh = (unsigned long long)(uintptr_t)cbh + (unsigned)tid * 16u;
        unsigned long long gl = (unsigned long long)(uintptr_t)cbl + (unsigned)tid * 16u;
        async_g2lds_b128(lds_off(&sh_h[0][0]) + (unsigned)tid * 16u, gh);
        async_g2lds_b128(lds_off(&sh_l[0][0]) + (unsigned)tid * 16u, gl);
        if (tid < VQ_CHUNK) {
            unsigned long long gn = (unsigned long long)(uintptr_t)cbn + (unsigned)tid * 4u;
            async_g2lds_b32(lds_off(&sh_n[0][0]) + (unsigned)tid * 4u, gn);
        }
    }

    int buf = 0;
    for (int chunk = 0; chunk < nchunk; ++chunk) {
        // Own async copies into sh_*[buf] have landed; barrier makes all
        // waves' copies visible AND confirms everyone is done reading the
        // buffer that the next stage will overwrite.
        wait_asynccnt0();
        __syncthreads();

        if (chunk + 1 < nchunk) {         // stage next chunk into other buffer
            const unsigned long long cbyte = (unsigned long long)(chunk + 1) * (VQ_CHUNK * 32ull);
            unsigned long long gh = (unsigned long long)(uintptr_t)cbh + cbyte + (unsigned)tid * 16u;
            unsigned long long gl = (unsigned long long)(uintptr_t)cbl + cbyte + (unsigned)tid * 16u;
            async_g2lds_b128(lds_off(&sh_h[buf ^ 1][0]) + (unsigned)tid * 16u, gh);
            async_g2lds_b128(lds_off(&sh_l[buf ^ 1][0]) + (unsigned)tid * 16u, gl);
            if (tid < VQ_CHUNK) {
                unsigned long long gn = (unsigned long long)(uintptr_t)cbn +
                                        (unsigned long long)((chunk + 1) * VQ_CHUNK + tid) * 4u;
                async_g2lds_b32(lds_off(&sh_n[buf ^ 1][0]) + (unsigned)tid * 4u, gn);
            }
        }

        const _Float16* lh = &sh_h[buf][0];
        const _Float16* ll = &sh_l[buf][0];
        const float*    ln = &sh_n[buf][0];
        const int c0 = chunk * VQ_CHUNK;

#pragma unroll
        for (int sub = 0; sub < VQ_CHUNK / 16; ++sub) {
            const int nl = sub * 16 + l;              // entry within chunk
            v16h b1, b2;
#pragma unroll
            for (int j = 0; j < 8; ++j) {
                b1[j] = b1[j + 8] = lh[nl * 16 + kb + j];
                b2[j] = b2[j + 8] = ll[nl * 16 + kb + j];
            }
            float cn = ln[nl];
            v8f acc = {};
            acc = __builtin_amdgcn_wmma_f32_16x16x32_f16(false, a1, false, b1,
                                                         (short)0, acc, false, false);
            acc = __builtin_amdgcn_wmma_f32_16x16x32_f16(false, a2, false, b2,
                                                         (short)0, acc, false, false);
            const int n = c0 + nl;
#pragma unroll
            for (int r = 0; r < 8; ++r) {
                float s = fmaf(-2.f, acc[r], cn);
                bool c = s < bestv[r];                // v_cndmask, no EXEC change
                bestv[r] = c ? s : bestv[r];
                besti[r] = c ? n : besti[r];
            }
        }
        buf ^= 1;
    }

    // Min-reduce across the 16 lanes of each half; prefer lower index on ties.
#pragma unroll
    for (int m = 8; m >= 1; m >>= 1) {
#pragma unroll
        for (int r = 0; r < 8; ++r) {
            float ov = __shfl_xor(bestv[r], m, 32);
            int   oi = __shfl_xor(besti[r], m, 32);
            bool c = (ov < bestv[r]) || (ov == bestv[r] && oi < besti[r]);
            bestv[r] = c ? ov : bestv[r];
            besti[r] = c ? oi : besti[r];
        }
    }
    if (wave_g < ntiles && l == 0) {
#pragma unroll
        for (int r = 0; r < 8; ++r) idx_out[row0 + half * 8 + r] = besti[r];
    }
}

// ---------------------------------------------------------------------------
// Gather zq in NCHW from idx + f32 codebook (exact values for the decoder).
// ---------------------------------------------------------------------------
__global__ void gather_zq(const int* __restrict__ idx, const float* __restrict__ cb,
                          float* __restrict__ zq, int Bn, int Hh, int Ww) {
    const int total = Bn * 16 * Hh * Ww;
    int t = blockIdx.x * blockDim.x + threadIdx.x;
    if (t >= total) return;
    int xq = t % Ww;
    int yq = (t / Ww) % Hh;
    int d  = (t / (Ww * Hh)) % 16;
    int b  = t / (Ww * Hh * 16);
    int n = (b * Hh + yq) * Ww + xq;
    zq[t] = cb[(size_t)idx[n] * 16 + d];
}

// ---------------------------------------------------------------------------
// Launch
// ---------------------------------------------------------------------------
extern "C" void kernel_launch(void* const* d_in, const int* in_sizes, int n_in,
                              void* d_out, int out_size, void* d_ws, size_t ws_size,
                              hipStream_t stream) {
    (void)in_sizes; (void)n_in; (void)out_size; (void)ws_size;
    const float* x    = (const float*)d_in[0];
    const float* ew1  = (const float*)d_in[1];
    const float* eb1  = (const float*)d_in[2];
    const float* ew2  = (const float*)d_in[3];
    const float* eb2  = (const float*)d_in[4];
    const float* ew3  = (const float*)d_in[5];
    const float* eb3  = (const float*)d_in[6];
    const float* dw1  = (const float*)d_in[7];
    const float* db1  = (const float*)d_in[8];
    const float* dw2  = (const float*)d_in[9];
    const float* db2  = (const float*)d_in[10];
    const float* dw3  = (const float*)d_in[11];
    const float* db3  = (const float*)d_in[12];
    const float* cbk  = (const float*)d_in[13];

    // Sizes
    const int B = 4;
    const int N_E1 = B * 4 * 126 * 126;   // 254016
    const int N_E2 = B * 8 * 124 * 124;   // 492032
    const int N_ZF = B * 122 * 122 * 16;  // 952576 (NHWC)
    const int N_ZQ = N_ZF;                // NCHW
    const int N_D1 = N_E2;
    const int N_D2 = N_E1;
    const int N_Y  = B * 3 * 128 * 128;   // 196608
    const int NLAT = B * 122 * 122;       // 59536 latent vectors
    const int KCB  = 8192;

    float* y_out   = (float*)d_out;
    int*   idx_out = (int*)d_out + N_Y;

    // Workspace carve (all offsets 32B-aligned)
    float* e1  = (float*)d_ws;
    float* e2  = e1 + N_E1;
    float* zf  = e2 + N_E2;
    float* zq  = zf + N_ZF;
    float* dc1 = zq + N_ZQ;
    float* dc2 = dc1 + N_D1;
    float* cbn = dc2 + N_D2;
    _Float16* cbh = (_Float16*)(cbn + KCB);
    _Float16* cbl = cbh + (size_t)KCB * 16;

    const int TB = 256;
    #define NB(n) (((n) + TB - 1) / TB)

    // Encoder
    enc_conv<3, 4, true,  false><<<NB(N_E1), TB, 0, stream>>>(x,  ew1, eb1, e1, B, 128, 128);
    enc_conv<4, 8, true,  false><<<NB(N_E2), TB, 0, stream>>>(e1, ew2, eb2, e2, B, 126, 126);
    enc_conv<8, 16, false, true><<<NB(N_ZF), TB, 0, stream>>>(e2, ew3, eb3, zf, B, 124, 124);

    // VQ
    cb_prep<<<NB(KCB), TB, 0, stream>>>(cbk, cbn, cbh, cbl, KCB);
    const int ntiles = NLAT / 16;         // 3721, exact
    vq_argmin<<<(ntiles + 7) / 8, TB, 0, stream>>>(zf, cbh, cbl, cbn, idx_out, ntiles, KCB);
    gather_zq<<<NB(N_ZQ), TB, 0, stream>>>(idx_out, cbk, zq, B, 122, 122);

    // Decoder
    dec_conv<16, 8, true ><<<NB(N_D1), TB, 0, stream>>>(zq,  dw1, db1, dc1, B, 122, 122);
    dec_conv<8,  4, true ><<<NB(N_D2), TB, 0, stream>>>(dc1, dw2, db2, dc2, B, 124, 124);
    dec_conv<4,  3, false><<<NB(N_Y),  TB, 0, stream>>>(dc2, dw3, db3, y_out, B, 126, 126);

    #undef NB
}
